// LSTM_Layer_2576980377766
// MI455X (gfx1250) — matrix operand order repaired and checked by reading, not compiled, not run on hardware
//
#include <hip/hip_runtime.h>
#include <hip/hip_bf16.h>

// ---------------------------------------------------------------------------
// LSTM layer: B=64, T=512, D=512, U=1024.  h_final [64,1024] f32 -> d_out.
// bf16 WMMA (v_wmma_f32_16x16x32_bf16) for x@Wx and h@Wh folded per step;
// async global->LDS weight staging (ASYNCcnt path); double-buffered LDS with
// one barrier per 64-K chunk; fused gate elementwise; 512 step launches.
// ---------------------------------------------------------------------------

typedef __bf16 v16bf __attribute__((ext_vector_type(16)));
typedef float  v8f   __attribute__((ext_vector_type(8)));

union ABFrag {
    v16bf v;
    uint4 q[2];
};

static constexpr int kB = 64;
static constexpr int kT = 512;
static constexpr int kD = 512;
static constexpr int kU = 1024;
static constexpr int kG = 4 * kU;  // 4096 gate columns

// ---- fp32 -> bf16 round-to-nearest-even (bit pattern) ----------------------
__device__ __forceinline__ unsigned short f32_to_bf16_rne(float f) {
    unsigned int u = __float_as_uint(f);
    unsigned int lsb = (u >> 16) & 1u;
    u += 0x7fffu + lsb;
    return (unsigned short)(u >> 16);
}

__device__ __forceinline__ float fast_sigmoid(float x) {
    return 1.0f / (1.0f + __expf(-x));
}
__device__ __forceinline__ float fast_tanh(float x) {
    float t = __expf(2.0f * x);
    return (t - 1.0f) / (t + 1.0f);
}

// ---- prologue kernels ------------------------------------------------------
__global__ void cvt_f32_to_bf16(const float* __restrict__ in,
                                unsigned short* __restrict__ out, int n) {
    int i = blockIdx.x * blockDim.x + threadIdx.x;
    if (i < n) out[i] = f32_to_bf16_rne(in[i]);
}

// W [K, N] fp32 row-major -> WT [N, K] bf16 row-major (B operand: contiguous K)
__global__ void transpose_f32_to_bf16(const float* __restrict__ W,
                                      unsigned short* __restrict__ WT,
                                      int K, int N) {
    int i = blockIdx.x * blockDim.x + threadIdx.x;
    if (i < K * N) {
        int k = i / N;
        int n = i - k * N;
        WT[(size_t)n * K + k] = f32_to_bf16_rne(W[i]);
    }
}

__global__ void copy_f32(const float* __restrict__ in, float* __restrict__ out, int n) {
    int i = blockIdx.x * blockDim.x + threadIdx.x;
    if (i < n) out[i] = in[i];
}

// ---- async 64B/lane global->LDS copy (GLOBAL_LOAD_ASYNC_TO_LDS_B128) ------
__device__ __forceinline__ void async_copy64(unsigned lds_dst,
                                             const unsigned short* src) {
    asm volatile(
        "global_load_async_to_lds_b128 %0, %1, off\n\t"
        "global_load_async_to_lds_b128 %0, %1, off offset:16\n\t"
        "global_load_async_to_lds_b128 %0, %1, off offset:32\n\t"
        "global_load_async_to_lds_b128 %0, %1, off offset:48"
        :
        : "v"(lds_dst), "v"(src)
        : "memory");
}
__device__ __forceinline__ void wait_async0() {
    asm volatile("s_wait_asynccnt 0x0" ::: "memory");
}

// ---- one GEMM phase: acc[g] += A(16-row tile) * WT(gate g cols) ------------
// Block = 128 threads (4 waves).  64-K chunks, double-buffered LDS,
// ONE barrier per chunk (WAR on a buffer is gated by the following barrier).
__device__ __forceinline__ void gemm_phase(
    const unsigned short* __restrict__ Arow,   // this lane's A row base (k=0)
    const unsigned short* __restrict__ WT,     // [4U, Kdim] bf16
    int Kdim, int u0, int tid,
    uint4 (*sB)[4][16][9],                     // [buf][gate][col][9xuint4]
    v8f* acc, int& pbuf) {
    const int lane = tid & 31;
    const int half = lane >> 4;   // 0: lanes 0-15, 1: lanes 16-31
    const int l    = lane & 15;
    // cooperative-fill decomposition: 128 threads -> (gate, col, 32K-half)
    const int fg = tid >> 5;          // 0..3  gate
    const int fn = (tid >> 1) & 15;   // 0..15 column within tile
    const int fh = tid & 1;           // 0..1  which 32-K half of 64-K chunk

    const unsigned short* wrow = WT + (size_t)(fg * kU + u0 + fn) * Kdim;

    for (int kb = 0; kb < Kdim; kb += 64) {
        const int p = pbuf;
        // async-stage B tiles: each lane DMAs 64 bytes straight into LDS
        const unsigned short* src = wrow + kb + fh * 32;
        unsigned lds_dst = (unsigned)(uintptr_t)(&sB[p][fg][fn][fh * 4]);
        async_copy64(lds_dst, src);
        __builtin_prefetch((const void*)(src + 64), 0, 0);  // next K chunk

        // A fragments for both 32-K sub-chunks (independent of LDS)
        ABFrag a0, a1;
        a0.q[0] = *(const uint4*)(Arow + kb + half * 8);
        a0.q[1] = *(const uint4*)(Arow + kb + half * 8 + 16);
        a1.q[0] = *(const uint4*)(Arow + kb + 32 + half * 8);
        a1.q[1] = *(const uint4*)(Arow + kb + 32 + half * 8 + 16);

        wait_async0();        // this wave's DMA into LDS is done
        __syncthreads();      // everyone's DMA done; prev buffer readers done

#pragma unroll
        for (int g = 0; g < 4; ++g) {
            ABFrag bm;
            bm.q[0] = sB[p][g][l][half];
            bm.q[1] = sB[p][g][l][2 + half];
            acc[g] = __builtin_amdgcn_wmma_f32_16x16x32_bf16(
                false, a0.v, false, bm.v, (short)0, acc[g], false, false);
        }
#pragma unroll
        for (int g = 0; g < 4; ++g) {
            ABFrag bm;
            bm.q[0] = sB[p][g][l][4 + half];
            bm.q[1] = sB[p][g][l][6 + half];
            acc[g] = __builtin_amdgcn_wmma_f32_16x16x32_bf16(
                false, a1.v, false, bm.v, (short)0, acc[g], false, false);
        }
        pbuf = p ^ 1;
    }
}

// ---- one LSTM timestep -----------------------------------------------------
// grid = 64 (unit chunks of 16), block = 128 (4 waves = 4 batch tiles of 16)
__global__ __launch_bounds__(128) void lstm_step_kernel(
    const unsigned short* __restrict__ x_bf,   // [B,T,D] bf16
    int t,
    const unsigned short* __restrict__ h_in,   // [B,U] bf16
    unsigned short* __restrict__ h_out,        // [B,U] bf16
    float* __restrict__ c,                     // [B,U] f32 (in-place)
    const unsigned short* __restrict__ WxT,    // [4U,D] bf16
    const unsigned short* __restrict__ WhT,    // [4U,U] bf16
    const float* __restrict__ bias,            // [4U]  f32
    float* __restrict__ h_f32) {               // [B,U] f32 (d_out)
    __shared__ uint4 sB[2][4][16][9];          // double buffer, 144B row pad

    const int tid  = threadIdx.x;
    const int lane = tid & 31;
    const int wave = tid >> 5;
    const int half = lane >> 4;
    const int l    = lane & 15;
    const int u0   = blockIdx.x * 16;   // unit chunk
    const int m0   = wave * 16;         // batch tile

    v8f acc[4];
#pragma unroll
    for (int g = 0; g < 4; ++g) acc[g] = (v8f)(0.0f);

    int pbuf = 0;
    // phase 1: x_t @ Wx   (K = D = 512)
    const unsigned short* xrow =
        x_bf + ((size_t)(m0 + l) * kT + (size_t)t) * kD;
    gemm_phase(xrow, WxT, kD, u0, tid, sB, acc, pbuf);

    // phase 2: h @ Wh     (K = U = 1024)  — buffer parity carries over
    const unsigned short* hrow = h_in + (size_t)(m0 + l) * kU;
    gemm_phase(hrow, WhT, kU, u0, tid, sB, acc, pbuf);

    // fused LSTM cell: lane owns column u = u0 + l, rows m0+half*8+e (e=0..7)
    const float bi = bias[0 * kU + u0 + l];
    const float bf = bias[1 * kU + u0 + l];
    const float bg = bias[2 * kU + u0 + l];
    const float bo = bias[3 * kU + u0 + l];
    const int m_base = m0 + half * 8;
#pragma unroll
    for (int e = 0; e < 8; ++e) {
        const size_t idx = (size_t)(m_base + e) * kU + u0 + l;
        float iv = fast_sigmoid(acc[0][e] + bi);
        float fv = fast_sigmoid(acc[1][e] + bf);
        float gv = fast_tanh(acc[2][e] + bg);
        float ov = fast_sigmoid(acc[3][e] + bo);
        float cn = fv * c[idx] + iv * gv;
        c[idx] = cn;
        float hn = ov * fast_tanh(cn);
        h_f32[idx] = hn;
        h_out[idx] = f32_to_bf16_rne(hn);
    }
}

// ---------------------------------------------------------------------------
extern "C" void kernel_launch(void* const* d_in, const int* in_sizes, int n_in,
                              void* d_out, int out_size, void* d_ws, size_t ws_size,
                              hipStream_t stream) {
    const float* x  = (const float*)d_in[0];  // [64,512,512]
    const float* h0 = (const float*)d_in[1];  // [64,1024]
    const float* c0 = (const float*)d_in[2];  // [64,1024]
    const float* Wx = (const float*)d_in[3];  // [512,4096]
    const float* Wh = (const float*)d_in[4];  // [1024,4096]
    const float* b  = (const float*)d_in[5];  // [4096]
    float* out = (float*)d_out;               // [64,1024]

    // workspace carve-out (~46.8 MB total)
    char* p = (char*)d_ws;
    auto carve = [&](size_t bytes) {
        void* r = (void*)p;
        p += (bytes + 255) & ~(size_t)255;
        return r;
    };
    unsigned short* x_bf = (unsigned short*)carve((size_t)kB * kT * kD * 2);
    unsigned short* WxT  = (unsigned short*)carve((size_t)kG * kD * 2);
    unsigned short* WhT  = (unsigned short*)carve((size_t)kG * kU * 2);
    unsigned short* hb0  = (unsigned short*)carve((size_t)kB * kU * 2);
    unsigned short* hb1  = (unsigned short*)carve((size_t)kB * kU * 2);
    float*          cbuf = (float*)carve((size_t)kB * kU * 4);

    // prologue: conversions / transposes
    {
        int n = kB * kT * kD;
        cvt_f32_to_bf16<<<(n + 255) / 256, 256, 0, stream>>>(x, x_bf, n);
    }
    {
        int n = kD * kG;
        transpose_f32_to_bf16<<<(n + 255) / 256, 256, 0, stream>>>(Wx, WxT, kD, kG);
    }
    {
        int n = kU * kG;
        transpose_f32_to_bf16<<<(n + 255) / 256, 256, 0, stream>>>(Wh, WhT, kU, kG);
    }
    {
        int n = kB * kU;
        cvt_f32_to_bf16<<<(n + 255) / 256, 256, 0, stream>>>(h0, hb0, n);
        copy_f32<<<(n + 255) / 256, 256, 0, stream>>>(c0, cbuf, n);
        copy_f32<<<(n + 255) / 256, 256, 0, stream>>>(h0, out, n);
    }

    // recurrence: 512 sequential step launches (graph-capture friendly)
    for (int t = 0; t < kT; ++t) {
        unsigned short* hin  = (t & 1) ? hb1 : hb0;
        unsigned short* hout = (t & 1) ? hb0 : hb1;
        lstm_step_kernel<<<kG / 16 / 4, 128, 0, stream>>>(
            x_bf, t, hin, hout, cbuf, WxT, WhT, b, out);
    }
}
